// LldCurveLoss_360777253404
// MI455X (gfx1250) — compile-verified, matrix operand order checked by reading, not run on hardware
//
#include <hip/hip_runtime.h>
#include <hip/hip_bf16.h>

typedef int v8i __attribute__((ext_vector_type(8)));

#define NC 8
#define EPS 1e-5f
#define HW 4096           // 64*64
#define NB 4
#define NCH 13

// ---------- helpers ----------
__device__ __forceinline__ float logsig(float x) {
    // log(sigmoid(x)) = min(x,0) - log1p(exp(-|x|))   (numerically stable)
    return fminf(x, 0.f) - log1pf(__expf(-fabsf(x)));
}
__device__ __forceinline__ float sigm(float x) {
    return 1.f / (1.f + __expf(-x));
}

// workspace layout (floats):
//  [0] focal_sum  [1] ign_sum  [2] off_se_sum  [3] fg_sum  [4] cls_sum
//  [8 + b*8 + k]  dice partials per batch: spg,spp,sgg,s2pg,s2pp,s2gg,msum (k=0..6)
//  [40 + (b*2+c)*4 + k] emb partials: pull,push,cs,cd
#define WS_DICE(b)   (8  + (b)*8)
#define WS_EMB(b,c)  (40 + ((b)*2+(c))*4)

// ---------- kernel 0: zero workspace ----------
__global__ void zero_ws(float* ws) { ws[threadIdx.x] = 0.f; }

// ---------- kernel 1: elementwise terms (focal/dice/off/cls) ----------
__global__ void elemwise(const float* __restrict__ outp,
                         const float* __restrict__ gtc,
                         const float* __restrict__ gox,
                         const float* __restrict__ goy,
                         const int*   __restrict__ ignm,
                         const int*   __restrict__ fgm,
                         const float* __restrict__ gtcls,
                         float* __restrict__ ws) {
    const int tid = blockIdx.x * blockDim.x + threadIdx.x;   // 0..16383
    const int b = tid >> 12;
    const int n = tid & (HW - 1);
    const size_t base = (size_t)b * NCH * HW;

    // ---- focal conf ----
    float x   = outp[base + n];                 // channel 0
    float g   = gtc[tid];
    float ign = (ignm[tid] > 0) ? 1.f : 0.f;
    float fg  = (fgm[tid]  > 0) ? 1.f : 0.f;
    float ls  = logsig(x), lns = logsig(-x);
    float bce = -(g * ls + (1.f - g) * lns);
    float pf  = __expf(-bce);
    float conf = -(30.f * g * ls + (1.f - g) * lns);   // LINE_W = 30
    float om  = 1.f - pf;
    float focal = om * om * conf * ign;                // GAMMA = 2

    // ---- dice partials ----
    float ps = sigm(x);
    float q = 1.f - ps, h = 1.f - g;

    // ---- offset ----
    float po1 = sigm(outp[base + 1 * HW + n]);
    float po2 = sigm(outp[base + 2 * HW + n]);
    float d1 = po1 - gox[tid], d2 = po2 - goy[tid];
    float se = (d1 * d1 + d2 * d2) * fg;

    // ---- cls BCE ----
    float clsacc = 0.f;
#pragma unroll
    for (int c = 0; c < NC; ++c) {
        float xl = outp[base + (5 + c) * HW + n];
        float gc = gtcls[((size_t)b * NC + c) * HW + n];
        clsacc += -(gc * logsig(xl) + (1.f - gc) * logsig(-xl));
    }

    float vals[12];
    vals[0] = focal;          vals[1] = ign;
    vals[2] = se;             vals[3] = fg;
    vals[4] = clsacc * fg;
    vals[5]  = ps * g * ign;  vals[6]  = ps * ps * ign;  vals[7]  = g * g * ign;
    vals[8]  = q * h * ign;   vals[9]  = q * q * ign;    vals[10] = h * h * ign;
    vals[11] = ign;

    // wave32 reduction per value
#pragma unroll
    for (int i = 0; i < 12; ++i) {
        float v = vals[i];
#pragma unroll
        for (int off = 16; off > 0; off >>= 1) v += __shfl_xor(v, off, 32);
        vals[i] = v;
    }
    if ((threadIdx.x & 31) == 0) {
        atomicAdd(ws + 0, vals[0]);
        atomicAdd(ws + 1, vals[1]);
        atomicAdd(ws + 2, vals[2]);
        atomicAdd(ws + 3, vals[3]);
        atomicAdd(ws + 4, vals[4]);
        float* d = ws + WS_DICE(b);   // whole block is within one batch
#pragma unroll
        for (int k = 0; k < 7; ++k) atomicAdd(d + k, vals[5 + k]);
    }
}

// ---------- kernel 2: pairwise embedding loss (WMMA IU8 for the 'same' masks) ----
// grid.x = NB * 2 * 256  (batch x channel x row-tile), 256 threads = 8 waves.
// Each wave sweeps 32 column tiles. Per 16x16 pair tile:
//   same_ij = A x B with one-hot(label)*mask operands  -> v_wmma_i32_16x16x64_iu8
//   pair_ij from two ballots; distances via VALU using C/D layout.
__global__ void pairwise(const float* __restrict__ outputs,
                         const int*   __restrict__ line_idx,
                         const int*   __restrict__ line_id,
                         const int*   __restrict__ fgm,
                         float* __restrict__ ws) {
    const int blk  = blockIdx.x;
    const int ti   = blk & 255;          // row tile
    const int c    = (blk >> 8) & 1;     // 0: emb (ch3/line_idx), 1: emb_id (ch4/line_id)
    const int b    = blk >> 9;           // batch
    const int lane = threadIdx.x & 31;
    const int wave = threadIdx.x >> 5;   // 0..7

    const float* p  = outputs + ((size_t)b * NCH + 3 + c) * HW;
    const int* lab  = (c == 0 ? line_idx : line_id) + (size_t)b * HW;
    const int* fg   = fgm + (size_t)b * HW;

    // ---- row-tile data: lanes 0..15 hold rows 0..15 ----
    const int rbase = ti * 16;
    float prow = 0.f; int grow = 0; int mrow = 0;
    if (lane < 16) {
        prow = p[rbase + lane];
        grow = lab[rbase + lane];
        mrow = fg[rbase + lane] > 0;
    }
    const unsigned rowmask = (unsigned)__ballot(lane < 16 && mrow);

    // A operand: 16x64 IU8, labels 0..7 => only K=0..7 (lanes 0..15, VGPR0/1) nonzero
    unsigned a0 = 0u, a1 = 0u;
    if (lane < 16 && mrow) {
        if (grow < 4) a0 = 1u << (grow * 8);
        else          a1 = 1u << ((grow - 4) * 8);
    }
    v8i A = {};
    A[0] = (int)a0; A[1] = (int)a1;

    // per-lane row registers for the accumulate phase (C/D layout: lanes>=16 -> M+8)
    const int rowBase = (lane >= 16) ? 8 : 0;
    float prm[8];
#pragma unroll
    for (int r = 0; r < 8; ++r) prm[r] = __shfl(prow, rowBase + r, 32);

    float pull = 0.f, push = 0.f, cs = 0.f, cd = 0.f;

    for (int tj = wave; tj < 256; tj += 8) {
        const int cbase = tj * 16;
        // prefetch next column tile of p into caches (global_prefetch_b8 path)
        if (tj + 8 < 256) __builtin_prefetch(&p[(tj + 8) * 16 + (lane & 15)], 0, 1);

        float pcol = 0.f; int gcol = 0; int mcol = 0;
        if (lane < 16) {
            pcol = p[cbase + lane];
            gcol = lab[cbase + lane];
            mcol = fg[cbase + lane] > 0;
        }
        const unsigned colmask = (unsigned)__ballot(lane < 16 && mcol);

        // B operand: 64x16 IU8, columns in lanes 0..15, K=0..7 in VGPR0/1
        unsigned b0 = 0u, b1 = 0u;
        if (lane < 16 && mcol) {
            if (gcol < 4) b0 = 1u << (gcol * 8);
            else          b1 = 1u << ((gcol - 4) * 8);
        }
        v8i Bm = {};
        Bm[0] = (int)b0; Bm[1] = (int)b1;
        v8i Cz = {};

        // same_ij tile: D = A x B  (16x16 i32)
        v8i D = __builtin_amdgcn_wmma_i32_16x16x64_iu8(false, A, false, Bm, Cz,
                                                       false, false);

        const int col = lane & 15;
        const float pc = __shfl(pcol, col, 32);
        const float mc = (float)((colmask >> col) & 1u);

#pragma unroll
        for (int r = 0; r < 8; ++r) {
            float same  = (float)D[r];
            float pairm = ((rowmask >> (rowBase + r)) & 1u) ? mc : 0.f;
            float d     = fabsf(prm[r] - pc);
            pull += fmaxf(d - 0.5f, 0.f) * same;       // PULL = 0.5
            float diff = pairm - same;
            push += fmaxf(1.f - d, 0.f) * diff;        // PUSH = 1.0
            cs += same;
            cd += diff;
        }
    }

    // wave32 reduction
#pragma unroll
    for (int off = 16; off > 0; off >>= 1) {
        pull += __shfl_xor(pull, off, 32);
        push += __shfl_xor(push, off, 32);
        cs   += __shfl_xor(cs,   off, 32);
        cd   += __shfl_xor(cd,   off, 32);
    }
    if (lane == 0) {
        float* e = ws + WS_EMB(b, c);
        atomicAdd(e + 0, pull);
        atomicAdd(e + 1, push);
        atomicAdd(e + 2, cs);
        atomicAdd(e + 3, cd);
    }
}

// ---------- kernel 3: combine everything into the scalar ----------
__global__ void finalize(const float* __restrict__ ws, float* __restrict__ out) {
    if (threadIdx.x != 0 || blockIdx.x != 0) return;
    float focal = ws[0], ign = ws[1], offs = ws[2], fg = ws[3], cls = ws[4];

    float conf_loss = focal / fmaxf(ign, 1.f);
    float dice = 0.f;
    for (int b = 0; b < NB; ++b) {
        const float* d = ws + WS_DICE(b);
        float spg = d[0], spp = d[1], sgg = d[2];
        float s2pg = d[3], s2pp = d[4], s2gg = d[5], ms = d[6];
        float t = 1.f - (spg + EPS) / (spp + sgg + EPS)
                      - (s2pg + EPS) / (s2pp + s2gg + EPS);
        dice += (ms > 0.f) ? t : 0.f;
    }
    conf_loss += 5.f * dice / (float)NB;               // DICE_W = 5

    float off_loss = offs / fmaxf(2.f * fg, 1.f);
    float cls_loss = cls  / fmaxf((float)NC * fg, 1.f);

    float emb = 0.f, embid = 0.f;
    for (int b = 0; b < NB; ++b)
        for (int c = 0; c < 2; ++c) {
            const float* e = ws + WS_EMB(b, c);
            float pull = e[0] / fmaxf(e[2], 1.f);
            float push = (e[3] > 0.f) ? e[1] / fmaxf(e[3], 1.f) : 0.f;
            if (c == 0) emb += pull + push; else embid += pull + push;
        }
    emb /= (float)NB; embid /= (float)NB;

    // CONF_W, OFF_W, EMB_W, EMBID_W, CLS_W = 1, 0.5, 1, 1, 1
    out[0] = conf_loss + 0.5f * off_loss + emb + embid + cls_loss;
}

extern "C" void kernel_launch(void* const* d_in, const int* in_sizes, int n_in,
                              void* d_out, int out_size, void* d_ws, size_t ws_size,
                              hipStream_t stream) {
    const float* outputs = (const float*)d_in[0];
    const float* gtc     = (const float*)d_in[1];
    const float* gox     = (const float*)d_in[2];
    const float* goy     = (const float*)d_in[3];
    const int*   lidx    = (const int*)d_in[4];
    const int*   ignm    = (const int*)d_in[5];
    const int*   fgm     = (const int*)d_in[6];
    const int*   lid     = (const int*)d_in[7];
    const float* gtcls   = (const float*)d_in[8];
    // d_in[9] (foreground_expand_mask) unused by the loss

    float* ws  = (float*)d_ws;
    float* out = (float*)d_out;

    zero_ws<<<1, 128, 0, stream>>>(ws);
    elemwise<<<(NB * HW) / 256, 256, 0, stream>>>(outputs, gtc, gox, goy,
                                                  ignm, fgm, gtcls, ws);
    pairwise<<<NB * 2 * 256, 256, 0, stream>>>(outputs, lidx, lid, fgm, ws);
    finalize<<<1, 32, 0, stream>>>(ws, out);
}
